// SynBlock_2473901162884
// MI455X (gfx1250) — compile-verified, matrix operand order checked by reading, not run on hardware
//
#include <hip/hip_runtime.h>
#include <hip/hip_bf16.h>
#include <math.h>

// ---------------------------------------------------------------------------
// CDNA5 (gfx1250) implementation of SynBlock forward.
// Conv / GEMM work routed through v_wmma_f32_16x16x32_f16 (wave32).
// Fast path for Cin%32==0 convs: tap hoisted to uniform outer loop so the
// K-chunk gather is immediate-offset loads only.
// ---------------------------------------------------------------------------

#define USE_ASYNC_LDS 1   // inline-asm global_load_async_to_lds_b32 staging

typedef __attribute__((ext_vector_type(16))) _Float16 v16h;
typedef __attribute__((ext_vector_type(8)))  float    v8f;

#define ACT_NONE  0
#define ACT_RELU  1
#define ACT_LEAKY 2
#define ACT_SIG   3

// K-map for 16-bit A/B fragments (16x32 per the CDNA5 ISA layout):
// lane l, vector element e  ->  K = ((e&8)<<1) + ((l&16)>>1) + (e&7)
__device__ __forceinline__ int wmma_kmap(int lane, int e) {
  return ((e & 8) << 1) + ((lane & 16) >> 1) + (e & 7);
}

__device__ __forceinline__ float conv_act(float v, int act) {
  if      (act == ACT_RELU)  v = v > 0.f ? v : 0.f;
  else if (act == ACT_LEAKY) v = v > 0.f ? v : 0.2f * v;
  else if (act == ACT_SIG)   v = 1.f / (1.f + expf(-v));
  return v;
}

// ---------------------------------------------------------------------------
// FAST implicit-GEMM conv: requires Cin % 32 == 0. One wave computes a
// 16-pixel x 32-out-channel tile (two 16x16 D tiles sharing one A fragment).
// Outer loop over taps (uniform ky/kx/iy), inner loop over channel chunks:
// all loads are base + compile-time-constant offsets (KS is a template arg).
// ---------------------------------------------------------------------------
template <int KS>
__global__ __launch_bounds__(32) void conv2d_wmma_fast(
    const float* __restrict__ in, const float* __restrict__ w,
    const float* __restrict__ bias, float* __restrict__ out,
    int N, int Cin, int Cout, int H, int W,
    int pad, int dil, int act, int cobase, int couttot)
{
  const int WSTR = KS * KS;                  // weight tap stride
  const int ntn = (Cout + 31) >> 5;
  const int ntm = (H * W) >> 4;
  int bid = blockIdx.x;
  const int tn = bid % ntn; bid /= ntn;
  const int tm = bid % ntm; const int n = bid / ntm;

  const int l    = threadIdx.x;
  const int mrow = l & 15;
  const int kl   = (l & 16) >> 1;

  const int pixb = tm << 4;
  const int y    = pixb / W;                 // uniform (W % 16 == 0)
  const int x    = (pixb % W) + mrow;        // per-lane A pixel

  const int  co0  = (tn << 5) + mrow;
  const int  co1  = co0 + 16;
  const bool two  = ((tn << 5) + 16) < Cout; // wave-uniform
  const int  co0c = min(co0, Cout - 1);
  const int  co1c = min(co1, Cout - 1);
  const float* __restrict__ wrow0 = w + (size_t)co0c * Cin * WSTR;
  const float* __restrict__ wrow1 = w + (size_t)co1c * Cin * WSTR;
  const float* __restrict__ inb   = in + (size_t)n * Cin * H * W;
  const size_t HWs = (size_t)H * W;

  v8f acc0 = {}, acc1 = {};
  for (int tap = 0; tap < WSTR; ++tap) {
    int iy = y, ix = x;
    if (KS == 3) {
      const int ky = tap / 3;                // tap is loop-invariant scalar
      const int kx = tap - ky * 3;
      iy = y + ky * dil - pad;               // uniform
      ix = x + kx * dil - pad;               // per-lane
    }
    const bool oka = (iy >= 0) & (iy < H) & (ix >= 0) & (ix < W);
    const int  iyc = min(max(iy, 0), H - 1);
    const int  ixc = min(max(ix, 0), W - 1);
    const float* __restrict__ ab = inb + (size_t)iyc * W + ixc;

    for (int cc = 0; cc < Cin; cc += 32) {
      const float* __restrict__ ap  = ab + (size_t)(cc + kl) * HWs;
      const float* __restrict__ wp0 = wrow0 + (size_t)(cc + kl) * WSTR + tap;
      __builtin_prefetch(ap + 32 * HWs);     // speculative -> global_prefetch_b8
      v16h af, bf0;
#pragma unroll
      for (int e = 0; e < 16; ++e) {
        const int ce = ((e & 8) << 1) + (e & 7);   // compile-time constant
        const float av = ap[(size_t)ce * HWs];
        af[e]  = (_Float16)(oka ? av : 0.f);
        bf0[e] = (_Float16)wp0[ce * WSTR];
      }
      acc0 = __builtin_amdgcn_wmma_f32_16x16x32_f16(
          false, af, false, bf0, (short)0, acc0, false, false);
      if (two) {
        const float* __restrict__ wp1 = wrow1 + (size_t)(cc + kl) * WSTR + tap;
        v16h bf1;
#pragma unroll
        for (int e = 0; e < 16; ++e) {
          const int ce = ((e & 8) << 1) + (e & 7);
          bf1[e] = (_Float16)wp1[ce * WSTR];
        }
        acc1 = __builtin_amdgcn_wmma_f32_16x16x32_f16(
            false, af, false, bf1, (short)0, acc1, false, false);
      }
    }
  }

  const int mhalf = (l & 16) >> 1;
  if (co0 < Cout) {
    const float bs = bias ? bias[co0] : 0.f;
    float* __restrict__ ob = out + ((size_t)n * couttot + cobase + co0) * HWs;
#pragma unroll
    for (int r = 0; r < 8; ++r)
      ob[pixb + r + mhalf] = conv_act(acc0[r] + bs, act);
  }
  if (two && co1 < Cout) {
    const float bs = bias ? bias[co1] : 0.f;
    float* __restrict__ ob = out + ((size_t)n * couttot + cobase + co1) * HWs;
#pragma unroll
    for (int r = 0; r < 8; ++r)
      ob[pixb + r + mhalf] = conv_act(acc1[r] + bs, act);
  }
}

// ---------------------------------------------------------------------------
// Generic implicit-GEMM conv (any Cin): K laid out as k = tap*Cpad + ci,
// Cpad = 2^cshift >= Cin; masked branchless gather.
// ---------------------------------------------------------------------------
__global__ __launch_bounds__(32) void conv2d_wmma(
    const float* __restrict__ in, const float* __restrict__ w,
    const float* __restrict__ bias, float* __restrict__ out,
    int N, int Cin, int Cout, int H, int W,
    int ks, int pad, int dil, int act, int cobase, int couttot, int cshift)
{
  const int Cpad = 1 << cshift;
  const int Kpad = (ks == 3) ? 9 * Cpad : Cpad;
  const int ntn = (Cout + 31) >> 5;
  const int ntm = (H * W) >> 4;
  int bid = blockIdx.x;
  const int tn = bid % ntn; bid /= ntn;
  const int tm = bid % ntm; const int n = bid / ntm;

  const int l    = threadIdx.x;
  const int mrow = l & 15;
  const int kl   = (l & 16) >> 1;

  const int pixb = tm << 4;
  const int y    = pixb / W;
  const int x    = (pixb % W) + mrow;

  const int  co0   = (tn << 5) + mrow;
  const int  co1   = co0 + 16;
  const bool two   = ((tn << 5) + 16) < Cout;
  const int  co0c  = min(co0, Cout - 1);
  const int  co1c  = min(co1, Cout - 1);
  const int  wrstr = Cin * ks * ks;
  const float* __restrict__ wrow0 = w + (size_t)co0c * wrstr;
  const float* __restrict__ wrow1 = w + (size_t)co1c * wrstr;
  const float* __restrict__ inb   = in + (size_t)n * Cin * H * W;

  v8f acc0 = {}, acc1 = {};
  for (int k0 = 0; k0 < Kpad; k0 += 32) {
    v16h af, bf0;
#pragma unroll
    for (int e = 0; e < 16; ++e) {
      const int kk  = k0 + ((e & 8) << 1) + kl + (e & 7);
      const int ci  = kk & (Cpad - 1);
      const int tap = kk >> cshift;
      const int cic = min(ci, Cin - 1);
      int iy, ix, widx;
      bool okk;
      if (ks == 3) {
        const int ky = tap / 3;
        const int kx = tap - ky * 3;
        iy = y + ky * dil - pad;
        ix = x + kx * dil - pad;
        okk  = (tap < 9) & (ci < Cin);
        widx = cic * 9 + min(tap, 8);
      } else {
        iy = y; ix = x;
        okk  = (ci < Cin);
        widx = cic;
      }
      const bool  oka = okk & (iy >= 0) & (iy < H) & (ix >= 0) & (ix < W);
      const int   iyc = min(max(iy, 0), H - 1);
      const int   ixc = min(max(ix, 0), W - 1);
      const float avl = inb[((size_t)cic * H + iyc) * W + ixc];
      const float bvl = wrow0[widx];
      af[e]  = (_Float16)(oka ? avl : 0.f);
      bf0[e] = (_Float16)(okk ? bvl : 0.f);
    }
    acc0 = __builtin_amdgcn_wmma_f32_16x16x32_f16(
        false, af, false, bf0, (short)0, acc0, false, false);
    if (two) {
      v16h bf1;
#pragma unroll
      for (int e = 0; e < 16; ++e) {
        const int kk  = k0 + ((e & 8) << 1) + kl + (e & 7);
        const int ci  = kk & (Cpad - 1);
        const int tap = kk >> cshift;
        const int cic = min(ci, Cin - 1);
        int widx; bool okk;
        if (ks == 3) { okk = (tap < 9) & (ci < Cin); widx = cic * 9 + min(tap, 8); }
        else         { okk = (ci < Cin);             widx = cic; }
        const float bvl = wrow1[widx];
        bf1[e] = (_Float16)(okk ? bvl : 0.f);
      }
      acc1 = __builtin_amdgcn_wmma_f32_16x16x32_f16(
          false, af, false, bf1, (short)0, acc1, false, false);
    }
  }

  const int mhalf = (l & 16) >> 1;
  if (co0 < Cout) {
    const float bs = bias ? bias[co0] : 0.f;
    float* __restrict__ ob = out + ((size_t)n * couttot + cobase + co0) * H * W;
#pragma unroll
    for (int r = 0; r < 8; ++r)
      ob[pixb + r + mhalf] = conv_act(acc0[r] + bs, act);
  }
  if (two && co1 < Cout) {
    const float bs = bias ? bias[co1] : 0.f;
    float* __restrict__ ob = out + ((size_t)n * couttot + cobase + co1) * H * W;
#pragma unroll
    for (int r = 0; r < 8; ++r)
      ob[pixb + r + mhalf] = conv_act(acc1[r] + bs, act);
  }
}

// ---------------------------------------------------------------------------
// corr[b,k,c] = sqrt(32) * sum_n e16[b,k,n] * f[b,c,n]   (M=16,N=32,K=16384)
// ---------------------------------------------------------------------------
__global__ __launch_bounds__(32) void corr_wmma(
    const float* __restrict__ e16, const float* __restrict__ f,
    float* __restrict__ corr)
{
  const int tn = blockIdx.x & 1;
  const int b  = blockIdx.x >> 1;
  const int l  = threadIdx.x;
  const int mrow = l & 15;
  const int c    = tn * 16 + mrow;
  const float* __restrict__ eb = e16 + (size_t)(b * 16 + mrow) * 16384;
  const float* __restrict__ fb = f   + (size_t)(b * 32 + c)    * 16384;

  v8f acc = {};
  for (int k0 = 0; k0 < 16384; k0 += 32) {
    v16h afrag, bfrag;
#pragma unroll
    for (int e = 0; e < 16; ++e) {
      const int kk = k0 + wmma_kmap(l, e);
      afrag[e] = (_Float16)eb[kk];
      bfrag[e] = (_Float16)fb[kk];
    }
    acc = __builtin_amdgcn_wmma_f32_16x16x32_f16(
        false, afrag, false, bfrag, (short)0, acc, false, false);
  }
  const int mhalf = (l & 16) >> 1;
#pragma unroll
  for (int r = 0; r < 8; ++r)
    corr[(b * 16 + r + mhalf) * 32 + c] = acc[r] * 5.656854249492381f;
}

// ---------------------------------------------------------------------------
// feav[b,k,n] = sum_c corr[b,k,c] * f[b,c,n]   (M=16, N=16 pixels/wave, K=32)
// ---------------------------------------------------------------------------
__global__ __launch_bounds__(32) void feav_wmma(
    const float* __restrict__ corr, const float* __restrict__ f,
    float* __restrict__ feav)
{
  const int tile = blockIdx.x & 1023;
  const int b    = blockIdx.x >> 10;
  const int l    = threadIdx.x;
  const int mrow = l & 15;
  const int pix  = tile * 16 + mrow;

  v16h afrag, bfrag;
#pragma unroll
  for (int e = 0; e < 16; ++e) {
    const int kk = wmma_kmap(l, e);
    afrag[e] = (_Float16)corr[(b * 16 + mrow) * 32 + kk];
    bfrag[e] = (_Float16)f[(size_t)(b * 32 + kk) * 16384 + pix];
  }
  v8f acc = {};
  acc = __builtin_amdgcn_wmma_f32_16x16x32_f16(
      false, afrag, false, bfrag, (short)0, acc, false, false);
  const int mhalf = (l & 16) >> 1;
#pragma unroll
  for (int r = 0; r < 8; ++r)
    feav[(size_t)(b * 16 + r + mhalf) * 16384 + tile * 16 + mrow] = acc[r];
}

// ------------------------- elementwise / small kernels ----------------------

__global__ void slice_fea_k(const float* __restrict__ fea,
                            float* __restrict__ out, int t)
{
  int idx = blockIdx.x * blockDim.x + threadIdx.x;
  if (idx >= 2 * 64 * 16384) return;
  const int p = idx & 16383;
  const int c = (idx >> 14) & 63;
  const int b = idx >> 20;
  out[idx] = fea[(((size_t)(b * 64 + c) * 4) + t) * 16384 + p];
}

__global__ void bilinear_k(const float* __restrict__ in, float* __restrict__ out,
                           int N, int C, int inH, int inW, int outH, int outW,
                           int align, long long inBase, long long inStride)
{
  int idx = blockIdx.x * blockDim.x + threadIdx.x;
  const int total = N * C * outH * outW;
  if (idx >= total) return;
  const int ox = idx % outW; int t = idx / outW;
  const int oy = t % outH;   t /= outH;
  const int c  = t % C;      const int n = t / C;

  float sy, sx;
  if (align) {
    sy = (outH > 1) ? (float)oy * (float)(inH - 1) / (float)(outH - 1) : 0.f;
    sx = (outW > 1) ? (float)ox * (float)(inW - 1) / (float)(outW - 1) : 0.f;
  } else {
    sy = fminf(fmaxf(((float)oy + 0.5f) * (float)inH / (float)outH - 0.5f, 0.f),
               (float)(inH - 1));
    sx = fminf(fmaxf(((float)ox + 0.5f) * (float)inW / (float)outW - 0.5f, 0.f),
               (float)(inW - 1));
  }
  const int y0 = (int)floorf(sy), y1 = min(y0 + 1, inH - 1);
  const int x0 = (int)floorf(sx), x1 = min(x0 + 1, inW - 1);
  const float wy = sy - (float)y0, wx = sx - (float)x0;
  const float* ib = in + inBase + (size_t)n * inStride + (size_t)c * inH * inW;
  const float v00 = ib[y0 * inW + x0], v01 = ib[y0 * inW + x1];
  const float v10 = ib[y1 * inW + x0], v11 = ib[y1 * inW + x1];
  const float top = v00 * (1.f - wx) + v01 * wx;
  const float bot = v10 * (1.f - wx) + v11 * wx;
  out[idx] = top * (1.f - wy) + bot * wy;
}

// Global average pool; streams through LDS via the gfx1250 async DMA path.
__global__ void gap_k(const float* __restrict__ in, float* __restrict__ out,
                      int HW)
{
  __shared__ float s[128];
  const int nc = blockIdx.x, t = threadIdx.x;
  const float* p = in + (size_t)nc * HW;
  float acc = 0.f;
#if USE_ASYNC_LDS
  for (int i0 = 0; i0 < HW; i0 += 128) {
    const unsigned ldsoff = t * 4u;
    const float*   g      = p + i0 + t;
    asm volatile("global_load_async_to_lds_b32 %0, %1, off"
                 :: "v"(ldsoff), "v"(g) : "memory");
    asm volatile("s_wait_asynccnt 0" ::: "memory");
    __syncthreads();
    acc += s[t];
    __syncthreads();
  }
#else
  for (int i = t; i < HW; i += 128) acc += p[i];
#endif
  s[t] = acc; __syncthreads();
  for (int o = 64; o > 0; o >>= 1) {
    if (t < o) s[t] += s[t + o];
    __syncthreads();
  }
  if (t == 0) out[nc] = s[0] / (float)HW;
}

__global__ void ca_mlp_k(const float* __restrict__ g,
                         const float* __restrict__ f1w, const float* __restrict__ f1b,
                         const float* __restrict__ f2w, const float* __restrict__ f2b,
                         float* __restrict__ att, int C, int R)
{
  __shared__ float a1[16];
  const int n = blockIdx.x, t = threadIdx.x;
  if (t < R) {
    float s = f1b[t];
    for (int c = 0; c < C; ++c) s += f1w[t * C + c] * g[n * C + c];
    a1[t] = s > 0.f ? s : 0.f;
  }
  __syncthreads();
  if (t < C) {
    float s = f2b[t];
    for (int r = 0; r < R; ++r) s += f2w[t * R + r] * a1[r];
    att[n * C + t] = 1.f / (1.f + expf(-s));
  }
}

__global__ void chan_scale_k(const float* __restrict__ in,
                             const float* __restrict__ att,
                             float* __restrict__ out, int total, int HW)
{
  int idx = blockIdx.x * blockDim.x + threadIdx.x;
  if (idx < total) out[idx] = in[idx] * att[idx / HW];
}

// ---------------------------------------------------------------------------
// Deformable separable conv + accumulate into out (which holds bias already).
// ---------------------------------------------------------------------------
__global__ void dsep_accum_k(const float* __restrict__ fr,
                             const float* __restrict__ v,  const float* __restrict__ h,
                             const float* __restrict__ ox, const float* __restrict__ oy,
                             const float* __restrict__ mk, float* __restrict__ out)
{
  const int H = 256, W = 256, HW = H * W;
  int idx = blockIdx.x * blockDim.x + threadIdx.x;
  if (idx >= 2 * HW) return;
  const int b = idx / HW, p = idx % HW;
  const int y = p / W, x = p % W;
  const float* fb = fr + (size_t)b * 3 * HW;

  float a0 = 0.f, a1 = 0.f, a2 = 0.f;
  for (int k = 0; k < 25; ++k) {
    const int kyi = k / 5, kxi = k % 5;
    const float py = oy[((size_t)(b * 25 + k)) * HW + p] + (float)(y + kyi);
    const float px = ox[((size_t)(b * 25 + k)) * HW + p] + (float)(x + kxi);
    const float wk = v[((size_t)(b * 5 + kyi)) * HW + p] *
                     h[((size_t)(b * 5 + kxi)) * HW + p] *
                     mk[((size_t)(b * 25 + k)) * HW + p];
    const float iy0 = floorf(py), ix0 = floorf(px);
    const float dy = py - iy0, dx = px - ix0;
#pragma unroll
    for (int cy = 0; cy < 2; ++cy) {
#pragma unroll
      for (int cx = 0; cx < 2; ++cx) {
        const float oyf = iy0 + (float)cy, oxf = ix0 + (float)cx;
        const float wy = cy ? dy : 1.f - dy;
        const float wx = cx ? dx : 1.f - dx;
        const bool valid = (oyf >= 0.f) & (oyf < 260.f) &
                           (oxf >= 0.f) & (oxf < 260.f);
        int sy = (int)oyf - 2; sy = sy < 0 ? 0 : (sy > 255 ? 255 : sy);
        int sx = (int)oxf - 2; sx = sx < 0 ? 0 : (sx > 255 ? 255 : sx);
        const float wt = valid ? wk * wy * wx : 0.f;
        const int si = sy * W + sx;
        a0 += wt * fb[si];
        a1 += wt * fb[HW + si];
        a2 += wt * fb[2 * HW + si];
      }
    }
  }
  out[((size_t)(b * 3 + 0)) * HW + p] += a0;
  out[((size_t)(b * 3 + 1)) * HW + p] += a1;
  out[((size_t)(b * 3 + 2)) * HW + p] += a2;
}

// ---------------------------------------------------------------------------
// Host orchestration
// ---------------------------------------------------------------------------
extern "C" void kernel_launch(void* const* d_in, const int* in_sizes, int n_in,
                              void* d_out, int out_size, void* d_ws, size_t ws_size,
                              hipStream_t stream)
{
  (void)in_sizes; (void)n_in; (void)out_size;
  const int CH = 128, CW = 128, H = 256, W = 256;
  const int HW128 = CH * CW;          // 16384
  const int HW    = H * W;            // 65536

  // Param leaf indices (insertion-order flattening of setup_inputs dict).
  const int P_FUSE_W = 3, P_FUSE_B = 4;
  const int HEAD_BIAS = 5, HEAD_V = 31, HEAD_H = 57, HEAD_OFFX = 83, HEAD_OFFY = 109;
  const int P_M1W = 135, P_M1B = 136, P_M2W = 137, P_M2B = 138;
  const int P_E1W = 139, P_E1B = 140, P_E2W = 141, P_E2B = 142;
  const int P_FDW = 143, P_FDB = 144, P_FUW = 145, P_FUB = 146;

  auto P = [&](int i) { return (const float*)d_in[i]; };
  const float* fea    = P(0);
  const float* frames = P(1);
  const float* edges  = P(2);
  float* outp = (float*)d_out;

  // ---- workspace carve-up (floats) ----
  float* ws = (float*)d_ws;
  size_t off = 0;
  auto alloc = [&](size_t n) { float* p = ws + off; off += n; return p; };
  float* bufA  = alloc((size_t)2 * 64 * HW128);
  float* bufB  = alloc((size_t)2 * 64 * HW128);
  float* bufU  = alloc((size_t)2 * 25 * HW);
  float* feat  = alloc((size_t)2 * 64 * HW128);
  float* v_t   = alloc((size_t)2 * 5  * HW);
  float* h_t   = alloc((size_t)2 * 5  * HW);
  float* ox_t  = alloc((size_t)2 * 25 * HW);
  float* oy_t  = alloc((size_t)2 * 25 * HW);
  float* m_t   = alloc((size_t)2 * 25 * HW);
  float* edget = alloc((size_t)2 * 1  * HW128);
  float* e8    = alloc((size_t)2 * 8  * HW128);
  float* e16   = alloc((size_t)2 * 16 * HW128);
  float* fbuf  = alloc((size_t)2 * 32 * HW128);
  float* feavb = alloc((size_t)2 * 16 * HW128);
  float* efb   = alloc((size_t)2 * 16 * HW128);
  float* corr  = alloc(1024);
  float* gbuf  = alloc(256);
  float* att   = alloc(256);
  if (off * sizeof(float) > ws_size) return;   // insufficient scratch

  auto log2ceil = [](int c) { int s = 0; while ((1 << s) < c) ++s; return s; };

  auto conv = [&](const float* in, int wi, int bi, float* out,
                  int N, int Cin, int Cout, int Hh, int Ww,
                  int ks, int pad, int dil, int act, int cobase, int couttot) {
    const int ntm = (Hh * Ww) >> 4;
    const int ntn = (Cout + 31) >> 5;
    const dim3 g((unsigned)(N * ntm * ntn));
    if ((Cin & 31) == 0) {
      if (ks == 3)
        conv2d_wmma_fast<3><<<g, 32, 0, stream>>>(
            in, P(wi), P(bi), out, N, Cin, Cout, Hh, Ww, pad, dil, act,
            cobase, couttot);
      else
        conv2d_wmma_fast<1><<<g, 32, 0, stream>>>(
            in, P(wi), P(bi), out, N, Cin, Cout, Hh, Ww, pad, dil, act,
            cobase, couttot);
    } else {
      conv2d_wmma<<<g, 32, 0, stream>>>(
          in, P(wi), P(bi), out, N, Cin, Cout, Hh, Ww, ks, pad, dil, act,
          cobase, couttot, log2ceil(Cin));
    }
  };
  auto launch1d = [&](int total) { return dim3((unsigned)((total + 255) / 256)); };

  // head(): PC3 -> CA -> PC2 -> CA -> conv -> up2x(align) -> conv
  auto headNet = [&](const float* xin, int base, int oc, float* outbuf) {
    // PC3 (relu fused, concat via channel offsets)
    conv(xin, base + 0,  base + 1,  bufB, 2, 64, 16, CH, CW, 3, 3, 3, ACT_RELU, 0,  64);
    conv(xin, base + 2,  base + 3,  bufB, 2, 64, 16, CH, CW, 3, 2, 2, ACT_RELU, 16, 64);
    conv(xin, base + 4,  base + 5,  bufB, 2, 64, 32, CH, CW, 3, 1, 1, ACT_RELU, 32, 64);
    // CA1
    conv(bufB, base + 6, base + 7,  bufA, 2, 64, 64, CH, CW, 1, 0, 1, ACT_RELU, 0, 64);
    gap_k<<<dim3(2 * 64), 128, 0, stream>>>(bufA, gbuf, HW128);
    ca_mlp_k<<<dim3(2), 64, 0, stream>>>(gbuf, P(base + 8), P(base + 9),
                                         P(base + 10), P(base + 11), att, 64, 4);
    chan_scale_k<<<launch1d(2 * 64 * HW128), 256, 0, stream>>>(
        bufA, att, bufB, 2 * 64 * HW128, HW128);
    // PC2
    conv(bufB, base + 12, base + 13, bufA, 2, 64, 16, CH, CW, 3, 2, 2, ACT_RELU, 0,  32);
    conv(bufB, base + 14, base + 15, bufA, 2, 64, 16, CH, CW, 3, 1, 1, ACT_RELU, 16, 32);
    // CA2
    conv(bufA, base + 16, base + 17, bufB, 2, 32, 32, CH, CW, 1, 0, 1, ACT_RELU, 0, 32);
    gap_k<<<dim3(2 * 32), 128, 0, stream>>>(bufB, gbuf, HW128);
    ca_mlp_k<<<dim3(2), 64, 0, stream>>>(gbuf, P(base + 18), P(base + 19),
                                         P(base + 20), P(base + 21), att, 32, 2);
    chan_scale_k<<<launch1d(2 * 32 * HW128), 256, 0, stream>>>(
        bufB, att, bufA, 2 * 32 * HW128, HW128);
    // ow conv (pad 1) + relu
    conv(bufA, base + 22, base + 23, bufB, 2, 32, oc, CH, CW, 3, 1, 1, ACT_RELU, 0, oc);
    // 2x bilinear upsample (align corners)
    bilinear_k<<<launch1d(2 * oc * HW), 256, 0, stream>>>(
        bufB, bufU, 2, oc, CH, CW, H, W, 1, 0, (long long)oc * CH * CW);
    // o2 conv (pad 1), no activation
    conv(bufU, base + 24, base + 25, outbuf, 2, oc, oc, H, W, 3, 1, 1, ACT_NONE, 0, oc);
  };

  // ---- Stage A: fuse (1x1, leaky 0.2) then BiasNet head -> d_out ----
  conv(fea, P_FUSE_W, P_FUSE_B, feat, 2, 256, 64, CH, CW, 1, 0, 1, ACT_LEAKY, 0, 64);
  headNet(feat, HEAD_BIAS, 3, outp);

  // ---- Stage B: per timestep ----
  for (int t = 0; t < 4; ++t) {
    slice_fea_k<<<launch1d(2 * 64 * HW128), 256, 0, stream>>>(fea, feat, t);
    bilinear_k<<<launch1d(2 * HW128), 256, 0, stream>>>(
        edges, edget, 2, 1, H, W, CH, CW, 0, (long long)t * HW, (long long)4 * HW);

    // edge fusion
    conv(edget, P_E1W, P_E1B, e8,   2, 1,  8,  CH, CW, 3, 1, 1, ACT_RELU, 0, 8);
    conv(e8,    P_E2W, P_E2B, e16,  2, 8,  16, CH, CW, 3, 1, 1, ACT_RELU, 0, 16);
    conv(feat,  P_FDW, P_FDB, fbuf, 2, 64, 32, CH, CW, 3, 1, 1, ACT_RELU, 0, 32);
    corr_wmma<<<dim3(4), 32, 0, stream>>>(e16, fbuf, corr);
    feav_wmma<<<dim3(2048), 32, 0, stream>>>(corr, fbuf, feavb);
    conv(feavb, P_FUW, P_FUB, efb, 2, 16, 16, CH, CW, 3, 1, 1, ACT_RELU, 0, 16);

    // kernel / offset heads
    headNet(feat, HEAD_V,    5,  v_t);
    headNet(feat, HEAD_H,    5,  h_t);
    headNet(feat, HEAD_OFFX, 25, ox_t);
    headNet(feat, HEAD_OFFY, 25, oy_t);

    // mask net
    conv(efb, P_M1W, P_M1B, bufA, 2, 16, 25, CH, CW, 3, 1, 1, ACT_RELU, 0, 25);
    bilinear_k<<<launch1d(2 * 25 * HW), 256, 0, stream>>>(
        bufA, bufU, 2, 25, CH, CW, H, W, 1, 0, (long long)25 * CH * CW);
    conv(bufU, P_M2W, P_M2B, m_t, 2, 25, 25, H, W, 3, 1, 1, ACT_SIG, 0, 25);

    // deformable separable conv (frames resize is identity at 256x256)
    const float* fr = frames + (size_t)t * 2 * 3 * HW;
    dsep_accum_k<<<launch1d(2 * HW), 256, 0, stream>>>(
        fr, v_t, h_t, ox_t, oy_t, m_t, outp);
  }
}
